// SmallIntegratedMoEModel_49125835931752
// MI455X (gfx1250) — matrix operand order repaired and checked by reading, not compiled, not run on hardware
//
#include <hip/hip_runtime.h>

// ---------------------------------------------------------------------------
// Types for CDNA5 WMMA (wave32): v_wmma_f32_16x16x32_bf16, and TDM descriptors
// ---------------------------------------------------------------------------
typedef __bf16 bf16_t;
typedef __attribute__((ext_vector_type(16))) __bf16 v16bf;
typedef __attribute__((ext_vector_type(8)))  float  v8f;
typedef __attribute__((ext_vector_type(4)))  unsigned int v4u;
typedef __attribute__((ext_vector_type(8)))  int v8i;
typedef __attribute__((ext_vector_type(4)))  int v4i;

__device__ __forceinline__ bf16_t f2bf(float f) {
  unsigned u = __float_as_uint(f);
  unsigned r = u + 0x7FFFu + ((u >> 16) & 1u);   // round-to-nearest-even
  unsigned short h = (unsigned short)(r >> 16);
  return __builtin_bit_cast(bf16_t, h);
}
__device__ __forceinline__ float leakyf(float v) { return v > 0.f ? v : 0.2f * v; }

// ---------------------------------------------------------------------------
// f32 -> bf16 elementwise converter (for MoE weights / activations)
// ---------------------------------------------------------------------------
__global__ void cvt_bf16_kernel(const float* __restrict__ in, bf16_t* __restrict__ out, int n) {
  int i = blockIdx.x * blockDim.x + threadIdx.x;
  if (i < n) out[i] = f2bf(in[i]);
}

// ---------------------------------------------------------------------------
// Direct conv + BN(eval) + ReLU  (scout router convs; tiny channel counts)
// ---------------------------------------------------------------------------
__global__ void conv_bn_relu_kernel(const float* __restrict__ in, const float* __restrict__ w,
                                    const float* __restrict__ bias,
                                    const float* __restrict__ g, const float* __restrict__ bb,
                                    const float* __restrict__ m, const float* __restrict__ v,
                                    float* __restrict__ out,
                                    int B, int Cin, int H, int W,
                                    int Cout, int KH, int KW, int stride, int pad,
                                    int Ho, int Wo) {
  int idx = blockIdx.x * blockDim.x + threadIdx.x;
  int total = B * Cout * Ho * Wo;
  if (idx >= total) return;
  int wo = idx % Wo; int t = idx / Wo;
  int ho = t % Ho;   t /= Ho;
  int co = t % Cout; int b = t / Cout;
  float acc = bias[co];
  int hi0 = ho * stride - pad, wi0 = wo * stride - pad;
  const float* wb = w + co * Cin * KH * KW;
  const float* ib = in + (b * Cin) * H * W;
  for (int ci = 0; ci < Cin; ci++)
    for (int kh = 0; kh < KH; kh++) {
      int hi = hi0 + kh; if ((unsigned)hi >= (unsigned)H) continue;
      const float* irow = ib + ci * H * W + hi * W;
      const float* wrow = wb + ci * KH * KW + kh * KW;
      for (int kw = 0; kw < KW; kw++) {
        int wi = wi0 + kw; if ((unsigned)wi >= (unsigned)W) continue;
        acc += irow[wi] * wrow[kw];
      }
    }
  float scale = g[co] * rsqrtf(v[co] + 1e-5f);
  float r = (acc - m[co]) * scale + bb[co];
  out[idx] = fmaxf(r, 0.f);
}

// ---------------------------------------------------------------------------
// Direct conv + leaky (trunk conv3)
// ---------------------------------------------------------------------------
__global__ void conv_leaky_kernel(const float* __restrict__ in, const float* __restrict__ w,
                                  const float* __restrict__ bias, float* __restrict__ out,
                                  int B, int Cin, int H, int W,
                                  int Cout, int KH, int KW, int stride, int pad,
                                  int Ho, int Wo) {
  int idx = blockIdx.x * blockDim.x + threadIdx.x;
  int total = B * Cout * Ho * Wo;
  if (idx >= total) return;
  int wo = idx % Wo; int t = idx / Wo;
  int ho = t % Ho;   t /= Ho;
  int co = t % Cout; int b = t / Cout;
  float acc = bias[co];
  int hi0 = ho * stride - pad, wi0 = wo * stride - pad;
  const float* wb = w + co * Cin * KH * KW;
  const float* ib = in + (b * Cin) * H * W;
  for (int ci = 0; ci < Cin; ci++)
    for (int kh = 0; kh < KH; kh++) {
      int hi = hi0 + kh; if ((unsigned)hi >= (unsigned)H) continue;
      const float* irow = ib + ci * H * W + hi * W;
      const float* wrow = wb + ci * KH * KW + kh * KW;
      for (int kw = 0; kw < KW; kw++) {
        int wi = wi0 + kw; if ((unsigned)wi >= (unsigned)W) continue;
        acc += irow[wi] * wrow[kw];
      }
    }
  out[idx] = leakyf(acc);
}

// ---------------------------------------------------------------------------
// Fused: conv + leaky + maxpool(2x2, stride 1).  Never materializes pre-pool
// tensor (saves ~250MB of HBM round-trips).  4x redundant conv is cheap here.
// ---------------------------------------------------------------------------
__global__ void conv_leaky_pool_kernel(const float* __restrict__ in, const float* __restrict__ w,
                                       const float* __restrict__ bias, float* __restrict__ out,
                                       int B, int Cin, int H, int W,
                                       int Cout, int KH, int KW, int stride, int pad,
                                       int HoP, int WoP) {
  int idx = blockIdx.x * blockDim.x + threadIdx.x;
  int total = B * Cout * HoP * WoP;
  if (idx >= total) return;
  int j = idx % WoP; int t = idx / WoP;
  int i = t % HoP;   t /= HoP;
  int co = t % Cout; int b = t / Cout;
  const float* wb = w + co * Cin * KH * KW;
  const float* ib = in + (b * Cin) * H * W;
  float best = -3.0e38f;
  for (int di = 0; di < 2; di++)
    for (int dj = 0; dj < 2; dj++) {
      int ho = i + di, wo = j + dj;
      float acc = bias[co];
      int hi0 = ho * stride - pad, wi0 = wo * stride - pad;
      for (int ci = 0; ci < Cin; ci++)
        for (int kh = 0; kh < KH; kh++) {
          int hi = hi0 + kh; if ((unsigned)hi >= (unsigned)H) continue;
          const float* irow = ib + ci * H * W + hi * W;
          const float* wrow = wb + ci * KH * KW + kh * KW;
          for (int kw = 0; kw < KW; kw++) {
            int wi = wi0 + kw; if ((unsigned)wi >= (unsigned)W) continue;
            acc += irow[wi] * wrow[kw];
          }
        }
      best = fmaxf(best, leakyf(acc));
    }
  out[idx] = best;
}

// ---------------------------------------------------------------------------
// Router: per-sample channel means (32ch x 28x28), logits (3), softmax,
// top-2 + renormalize -> comb_w[b][e].  One wave32 per sample; LDS reduce.
// ---------------------------------------------------------------------------
__global__ void router_kernel(const float* __restrict__ h2, const float* __restrict__ cw,
                              const float* __restrict__ cb, float* __restrict__ combw) {
  int b = blockIdx.x;
  int c = threadIdx.x;              // 0..31, one channel per lane
  const float* p = h2 + ((size_t)b * 32 + c) * 784;
  float s = 0.f;
  for (int i = 0; i < 784; i++) s += p[i];
  __shared__ float mean[32];
  mean[c] = s * (1.f / 784.f);
  __syncthreads();
  if (c == 0) {
    float logit[3];
    for (int e = 0; e < 3; e++) {
      float a = cb[e];
      for (int k = 0; k < 32; k++) a += mean[k] * cw[e * 32 + k];
      logit[e] = a;
    }
    float mx = fmaxf(logit[0], fmaxf(logit[1], logit[2]));
    float ex[3]; float sum = 0.f;
    for (int e = 0; e < 3; e++) { ex[e] = expf(logit[e] - mx); sum += ex[e]; }
    float pr[3];
    for (int e = 0; e < 3; e++) pr[e] = ex[e] / sum;
    int i1 = 0;
    for (int e = 1; e < 3; e++) if (pr[e] > pr[i1]) i1 = e;   // ties -> lower idx
    int i2 = -1;
    for (int e = 0; e < 3; e++) if (e != i1 && (i2 < 0 || pr[e] > pr[i2])) i2 = e;
    float denom = pr[i1] + pr[i2] + 1e-6f;
    for (int e = 0; e < 3; e++) {
      float wv = 0.f;
      if (e == i1) wv = pr[i1] / denom;
      else if (e == i2) wv = pr[i2] / denom;
      combw[b * 3 + e] = wv;
    }
  }
}

// ---------------------------------------------------------------------------
// Adaptive avg-pool 26x26 -> 3x3 (torch window rule) + flatten + bf16 cast
// ---------------------------------------------------------------------------
__global__ void adaptive_pool_flatten_kernel(const float* __restrict__ in, bf16_t* __restrict__ flat,
                                             int B, int C, int H, int W) {
  int idx = blockIdx.x * blockDim.x + threadIdx.x;
  int total = B * C * 9;
  if (idx >= total) return;
  int j = idx % 3; int t = idx / 3;
  int i = t % 3;   t /= 3;
  int c = t % C;   int b = t / C;
  int hs = (i * H) / 3, he = ((i + 1) * H + 2) / 3;
  int ws = (j * W) / 3, we = ((j + 1) * W + 2) / 3;
  const float* p = in + ((size_t)(b * C + c)) * H * W;
  float s = 0.f;
  for (int y = hs; y < he; y++)
    for (int x = ws; x < we; x++) s += p[y * W + x];
  float mean = s / (float)((he - hs) * (we - ws));
  flat[(size_t)b * (C * 9) + c * 9 + i * 3 + j] = f2bf(mean);
}

// ---------------------------------------------------------------------------
// WMMA bf16 GEMM:  out[M,N] = act( A[M,K] @ W[N,K]^T + bias[N] )
// A,W bf16 row-major; acc f32.  One wave per 16x16 tile, 4 waves/block
// sharing one A-panel (16 x K) staged into LDS by the Tensor Data Mover
// (tensor_load_to_lds + s_wait_tensorcnt).  B is streamed from L2 with a
// register double-buffer; the last K-iteration is peeled so the steady-state
// loop body unconditionally issues next-tile global_load_b128 before the
// current WMMA (no fallback register copies / extra hazard NOPs).
// ---------------------------------------------------------------------------
#define MAX_K 1152

template <bool OUT_BF16, bool LEAKY>
__global__ void gemm_bf16_wmma_kernel(const bf16_t* __restrict__ A, const bf16_t* __restrict__ W,
                                      const float* __restrict__ bias, void* __restrict__ out,
                                      int M, int N, int K) {
  __shared__ bf16_t ldsA[16 * MAX_K];
  const int lane  = threadIdx.x;                      // 0..31
  const int half  = lane >> 4;                        // lane group
  const int l     = lane & 15;
  const int ntile = blockIdx.x * blockDim.y + threadIdx.y;
  const int mtile = blockIdx.y;
  const int m0 = mtile * 16, n0 = ntile * 16;

#if __has_builtin(__builtin_amdgcn_tensor_load_to_lds) && defined(__gfx1250__)
  // ---- TDM: stage A panel (16 rows x K) into LDS with one DMA descriptor ---
  if (threadIdx.y == 0) {
    unsigned lds_off = (unsigned)(size_t)(void*)&ldsA[0];          // LDS byte offset
    unsigned long long ga = (unsigned long long)(const void*)(A + (size_t)m0 * K);
    // D# group 0: count=1 (valid), lds_addr, global_addr[56:0], type=2
    v4u g0 = { 1u,
               lds_off,
               (unsigned)(ga & 0xFFFFFFFFu),
               (unsigned)((ga >> 32) & 0x1FFFFFFu) | (2u << 30) };
    // D# group 1: data_size=1 (2B); tensor_dim0=K @bits79:48; tensor_dim1=16
    // @111:80; tile_dim0=K @127:112; tile_dim1=16 @143:128; dim0_stride=K @207:160
    v8i g1 = { (int)(1u << 16),
               (int)((unsigned)K << 16),
               (int)(16u << 16),
               (int)((unsigned)K << 16),
               16,
               K,
               0, 0 };
    v4i z4 = {0, 0, 0, 0};
    v8i z8 = {0, 0, 0, 0, 0, 0, 0, 0};
    __builtin_amdgcn_tensor_load_to_lds(g0, g1, z4, z4, z8, 0);
    __builtin_amdgcn_s_wait_tensorcnt(0);
  }
  __syncthreads();
#else
  // Fallback (also what the host pass parses): cooperative panel staging.
  for (int idx = threadIdx.y * 32 + lane; idx < 2 * K; idx += 128)
    *(uint4*)(&ldsA[idx * 8]) = *(const uint4*)(A + (size_t)m0 * K + idx * 8);
  __syncthreads();
#endif

  v8f acc = {};
  const bf16_t* wrow = W + (size_t)(n0 + l) * K;      // B column n = weight row n
  const bf16_t* lrow = ldsA + l * K;                  // A row l of the LDS panel

  union Frag { uint4 u[2]; v16bf v; };
  // Preload B fragment for k0 = 0 (B 32x16: half 0 -> K{0..7,16..23}, half 1 -> K{8..15,24..31})
  Frag bcur;
  bcur.u[0] = *(const uint4*)(wrow + 8 * half);
  bcur.u[1] = *(const uint4*)(wrow + 16 + 8 * half);

  int k0 = 0;
  // Steady state: unconditionally issue loads for tile k0+32, then WMMA tile k0.
  for (; k0 + 32 < K; k0 += 32) {
    Frag bnext;
    bnext.u[0] = *(const uint4*)(wrow + k0 + 32 + 8 * half);
    bnext.u[1] = *(const uint4*)(wrow + k0 + 32 + 16 + 8 * half);
    __builtin_prefetch(wrow + k0 + 96, 0, 3);         // gfx1250 global_prefetch_b8
    // A 16x32 from LDS: half 0 -> K {0..7,16..23}; half 1 -> K {8..15,24..31}
    Frag afrag;
    afrag.u[0] = *(const uint4*)(lrow + k0 + 8 * half);
    afrag.u[1] = *(const uint4*)(lrow + k0 + 16 + 8 * half);
    acc = __builtin_amdgcn_wmma_f32_16x16x32_bf16(
        /*neg_a=*/false, afrag.v, /*neg_b=*/false, bcur.v,
        /*c_mod=*/(short)0, acc, /*reuse_a=*/false, /*reuse_b=*/false);
    bcur = bnext;
  }
  // Peeled last iteration (K is a multiple of 32).
  {
    Frag afrag;
    afrag.u[0] = *(const uint4*)(lrow + k0 + 8 * half);
    afrag.u[1] = *(const uint4*)(lrow + k0 + 16 + 8 * half);
    acc = __builtin_amdgcn_wmma_f32_16x16x32_bf16(
        false, afrag.v, false, bcur.v, (short)0, acc, false, false);
  }

  const float bn = bias[n0 + l];
  const int col = n0 + l;
#pragma unroll
  for (int r = 0; r < 8; r++) {
    float val = acc[r] + bn;
    if (LEAKY) val = leakyf(val);
    const int row = m0 + r + 8 * half;                // C/D layout: M = vgpr + 8*half
    if (OUT_BF16) ((bf16_t*)out)[(size_t)row * N + col] = f2bf(val);
    else          ((float*)out)[(size_t)row * N + col] = val;
  }
}

// ---------------------------------------------------------------------------
// Final combine: out[b,o] = sum_e combw[b,e] * h3[e][b][o]
// ---------------------------------------------------------------------------
__global__ void combine_kernel(const float* __restrict__ h3, const float* __restrict__ combw,
                               float* __restrict__ out, int B, int F) {
  int idx = blockIdx.x * blockDim.x + threadIdx.x;
  if (idx >= B * F) return;
  int o = idx % F, b = idx / F;
  float s = 0.f;
  for (int e = 0; e < 3; e++)
    s += combw[b * 3 + e] * h3[((size_t)e * B + b) * F + o];
  out[idx] = s;
}

// ---------------------------------------------------------------------------
// Host launcher
// ---------------------------------------------------------------------------
static inline int cdiv(int a, int b) { return (a + b - 1) / b; }

extern "C" void kernel_launch(void* const* d_in, const int* in_sizes, int n_in,
                              void* d_out, int out_size, void* d_ws, size_t ws_size,
                              hipStream_t stream) {
  const float* x    = (const float*)d_in[0];
  const float* tw1  = (const float*)d_in[1];  const float* tb1 = (const float*)d_in[2];
  const float* tw2  = (const float*)d_in[3];  const float* tb2 = (const float*)d_in[4];
  const float* tw3  = (const float*)d_in[5];  const float* tb3 = (const float*)d_in[6];
  const float* tw4  = (const float*)d_in[7];  const float* tb4 = (const float*)d_in[8];
  const float* sw1  = (const float*)d_in[9];  const float* sb1 = (const float*)d_in[10];
  const float* bn1g = (const float*)d_in[11]; const float* bn1b = (const float*)d_in[12];
  const float* bn1m = (const float*)d_in[13]; const float* bn1v = (const float*)d_in[14];
  const float* sw2  = (const float*)d_in[15]; const float* sb2 = (const float*)d_in[16];
  const float* bn2g = (const float*)d_in[17]; const float* bn2b = (const float*)d_in[18];
  const float* bn2m = (const float*)d_in[19]; const float* bn2v = (const float*)d_in[20];
  const float* cw   = (const float*)d_in[21]; const float* cb  = (const float*)d_in[22];
  const float* ew1  = (const float*)d_in[23]; const float* eb1 = (const float*)d_in[24];
  const float* ew2  = (const float*)d_in[25]; const float* eb2 = (const float*)d_in[26];
  const float* ew3  = (const float*)d_in[27]; const float* eb3 = (const float*)d_in[28];

  const int B = 256;
  char* ws = (char*)d_ws;

  // Region plan (lifetime-based reuse; all sizes already 256B multiples):
  // A (84.5MB): scoutC1[256,16,56,56] -> P4[256,128,26,26]
  // B (47.3MB): scoutC2[256,32,28,28] -> P1[256,16,55,55] -> T3[256,64,27,27]
  // C (22.8MB): P2[256,32,27,27]
  const size_t offA = 0;
  const size_t offB = offA + (size_t)256 * 128 * 26 * 26 * 4;
  const size_t offC = offB + (size_t)256 * 16 * 55 * 55 * 4;
  size_t off = offC + (size_t)256 * 32 * 27 * 27 * 4;

  bf16_t* ew1b = (bf16_t*)(ws + off); off += (size_t)3 * 1024 * 1152 * 2;
  bf16_t* ew2b = (bf16_t*)(ws + off); off += (size_t)3 * 512 * 1024 * 2;
  bf16_t* ew3b = (bf16_t*)(ws + off); off += (size_t)3 * 512 * 512 * 2;
  bf16_t* flatb= (bf16_t*)(ws + off); off += (size_t)256 * 1152 * 2;
  bf16_t* h1b  = (bf16_t*)(ws + off); off += (size_t)256 * 1024 * 2;
  bf16_t* h2b  = (bf16_t*)(ws + off); off += (size_t)256 * 512 * 2;
  float*  h3f  = (float*)(ws + off);  off += (size_t)3 * 256 * 512 * 4;
  float*  combw= (float*)(ws + off);  off += (size_t)256 * 3 * 4;
  (void)ws_size; (void)in_sizes; (void)n_in; (void)out_size;

  float* RA = (float*)(ws + offA);
  float* RB = (float*)(ws + offB);
  float* RC = (float*)(ws + offC);

  const int TPB = 256;

  // --- MoE weight conversion f32 -> bf16 (independent) ---
  cvt_bf16_kernel<<<cdiv(3 * 1024 * 1152, TPB), TPB, 0, stream>>>(ew1, ew1b, 3 * 1024 * 1152);
  cvt_bf16_kernel<<<cdiv(3 * 512 * 1024, TPB), TPB, 0, stream>>>(ew2, ew2b, 3 * 512 * 1024);
  cvt_bf16_kernel<<<cdiv(3 * 512 * 512, TPB), TPB, 0, stream>>>(ew3, ew3b, 3 * 512 * 512);

  // --- Scout router ---
  conv_bn_relu_kernel<<<cdiv(B * 16 * 56 * 56, TPB), TPB, 0, stream>>>(
      x, sw1, sb1, bn1g, bn1b, bn1m, bn1v, RA, B, 3, 224, 224, 16, 7, 7, 4, 3, 56, 56);
  conv_bn_relu_kernel<<<cdiv(B * 32 * 28 * 28, TPB), TPB, 0, stream>>>(
      RA, sw2, sb2, bn2g, bn2b, bn2m, bn2v, RB, B, 16, 56, 56, 32, 3, 3, 2, 1, 28, 28);
  router_kernel<<<B, 32, 0, stream>>>(RB, cw, cb, combw);

  // --- Trunk ---
  conv_leaky_pool_kernel<<<cdiv(B * 16 * 55 * 55, TPB), TPB, 0, stream>>>(
      x, tw1, tb1, RB, B, 3, 224, 224, 16, 3, 3, 4, 1, 55, 55);
  conv_leaky_pool_kernel<<<cdiv(B * 32 * 27 * 27, TPB), TPB, 0, stream>>>(
      RB, tw2, tb2, RC, B, 16, 55, 55, 32, 5, 5, 2, 2, 27, 27);
  conv_leaky_kernel<<<cdiv(B * 64 * 27 * 27, TPB), TPB, 0, stream>>>(
      RC, tw3, tb3, RB, B, 32, 27, 27, 64, 3, 3, 1, 1, 27, 27);
  conv_leaky_pool_kernel<<<cdiv(B * 128 * 26 * 26, TPB), TPB, 0, stream>>>(
      RB, tw4, tb4, RA, B, 64, 27, 27, 128, 3, 3, 1, 1, 26, 26);
  adaptive_pool_flatten_kernel<<<cdiv(B * 128 * 9, TPB), TPB, 0, stream>>>(
      RA, flatb, B, 128, 26, 26);

  // --- MoE head: per-expert WMMA GEMM chain ---
  dim3 blk(32, 4, 1);
  for (int e = 0; e < 3; e++) {
    gemm_bf16_wmma_kernel<true, true><<<dim3(1024 / 64, 256 / 16), blk, 0, stream>>>(
        flatb, ew1b + (size_t)e * 1024 * 1152, eb1 + e * 1024, h1b, 256, 1024, 1152);
    gemm_bf16_wmma_kernel<true, true><<<dim3(512 / 64, 256 / 16), blk, 0, stream>>>(
        h1b, ew2b + (size_t)e * 512 * 1024, eb2 + e * 512, h2b, 256, 512, 1024);
    gemm_bf16_wmma_kernel<false, false><<<dim3(512 / 64, 256 / 16), blk, 0, stream>>>(
        h2b, ew3b + (size_t)e * 512 * 512, eb3 + e * 512, h3f + (size_t)e * 256 * 512,
        256, 512, 512);
  }

  // --- weighted combine -> d_out [256,512] f32 ---
  combine_kernel<<<cdiv(256 * 512, TPB), TPB, 0, stream>>>(h3f, combw, (float*)d_out, 256, 512);
}
